// ENN_23579370455608
// MI455X (gfx1250) — matrix operand order repaired and checked by reading, not compiled, not run on hardware
//
#include <hip/hip_runtime.h>
#include <math.h>

// ---------------------------------------------------------------------------
// CDNA5 (gfx1250) implementation of the neural-plasticity return-mapping scan.
// One wave32 owns 16 batch elements; every 30-wide MLP matmul (forward and
// backprop) is two V_WMMA_F32_16X16X32_F16 ops. Per-element 5x5 Newton solve
// runs on lanes 0..15 in fp32.  Weight fragments are fetched with true
// global_load_b128 (addrspace(1)) so FLAT ops don't couple LOADcnt with the
// wave's dense DScnt traffic.
// ---------------------------------------------------------------------------

typedef __attribute__((ext_vector_type(16))) _Float16 v16h;
typedef __attribute__((ext_vector_type(8)))  _Float16 v8h;
typedef __attribute__((ext_vector_type(8)))  float    v8f;

#define FD_EPS 1e-3f

// ---------------- weight-fragment tile indices ----------------
enum {
  T_W1_0 = 0, T_W1_1, T_S1_0, T_S1_1, T_A2_0, T_A2_1, T_A3_0, T_A3_1,
  T_A4_0, T_S3_0, T_Q1_0, T_Q1_1, T_Q2_0, T_Q2_1, T_Q3_0, T_Q3_1,
  T_Q4_0, T_A3T_0, T_A3T_1, T_A2T_0, T_A2T_1, T_S2_0, T_S2_1,
  T_S2T_0, T_S1T_0, T_W1T_0, T_Q3T_0, T_Q3T_1, T_Q2T_0, T_Q2T_1,
  T_Q1T_0, T_E0_0, T_E0_1, T_E1_0, T_E1_1, T_E2_0, T_E2_1, T_E3_0,
  T_E2T_0, T_E2T_1, T_E1T_0, T_E1T_1, T_E0T_0, NUM_TILES
};

// packed param-block offsets (floats)
enum { P_Z1B = 0, P_EB0 = 32, P_EB1 = 64, P_EB2 = 96, P_EB3 = 128,
       P_A4V = 132, P_Q4V = 164, P_E3V = 196, P_SZ3 = 228, PAR_N = 256 };

struct NetParams {
  const float *z1_w, *z1_b, *z2, *z3, *z4, *sz1, *sz2, *sz3;
  const float *q1, *q2, *q3, *q4;
  const float *e_w0, *e_b0, *e_w1, *e_b1, *e_w2, *e_b2, *e_w3, *e_b3;
};

// ---------------- prep kernel: pack B-fragments ----------------
// B-fragment layout assumption (consistent pack/load, mirrors the ISA's
// K-split across lane halves): lane L holds column n = L&15 of its N-tile,
// K = (L<16 ? {0..7,16..23} : {8..15,24..31}), 16 halves per lane.
#define TD(src, r, c, nt, tr, ab) \
  ((src) | ((r) << 4) | ((c) << 9) | ((nt) << 14) | ((tr) << 15) | ((ab) << 16))

__global__ void pack_frags(NetParams P, _Float16* __restrict__ WF) {
  const int idx = blockIdx.x * blockDim.x + threadIdx.x;
  if (idx >= NUM_TILES * 32) return;
  const int tile = idx >> 5, lane = idx & 31;
  const float* mats[15] = { P.z1_w, P.sz1, P.z2, P.sz2, P.z3, P.z4, P.sz3,
                            P.q1, P.q2, P.q3, P.q4, P.e_w0, P.e_w1, P.e_w2, P.e_w3 };
  const int D[NUM_TILES] = {
    TD(0,4,30,0,0,0),  TD(0,4,30,1,0,0),   // W1 = z1_w
    TD(1,4,30,0,0,0),  TD(1,4,30,1,0,0),   // S1 = sz1
    TD(2,30,30,0,0,1), TD(2,30,30,1,0,1),  // A2 = |z2|
    TD(4,30,30,0,0,1), TD(4,30,30,1,0,1),  // A3 = |z3|
    TD(5,30,1,0,0,1),                      // A4 = |z4|
    TD(6,4,1,0,0,0),                       // S3 = sz3
    TD(7,5,30,0,0,0),  TD(7,5,30,1,0,0),   // Q1
    TD(8,30,30,0,0,0), TD(8,30,30,1,0,0),  // Q2
    TD(9,30,30,0,0,0), TD(9,30,30,1,0,0),  // Q3
    TD(10,30,1,0,0,0),                     // Q4
    TD(4,30,30,0,1,1), TD(4,30,30,1,1,1),  // A3^T
    TD(2,30,30,0,1,1), TD(2,30,30,1,1,1),  // A2^T
    TD(3,4,30,0,0,0),  TD(3,4,30,1,0,0),   // S2 = sz2
    TD(3,4,30,0,1,0),                      // S2^T
    TD(1,4,30,0,1,0),                      // S1^T
    TD(0,4,30,0,1,0),                      // W1^T
    TD(9,30,30,0,1,0), TD(9,30,30,1,1,0),  // Q3^T
    TD(8,30,30,0,1,0), TD(8,30,30,1,1,0),  // Q2^T
    TD(7,5,30,0,1,0),                      // Q1^T
    TD(11,9,30,0,0,0), TD(11,9,30,1,0,0),  // E0
    TD(12,30,30,0,0,0),TD(12,30,30,1,0,0), // E1
    TD(13,30,30,0,0,0),TD(13,30,30,1,0,0), // E2
    TD(14,30,1,0,0,0),                     // E3
    TD(13,30,30,0,1,0),TD(13,30,30,1,1,0), // E2^T
    TD(12,30,30,0,1,0),TD(12,30,30,1,1,0), // E1^T
    TD(11,9,30,0,1,0)                      // E0^T
  };
  const int d = D[tile];
  const int src = d & 15, R = (d >> 4) & 31, C = (d >> 9) & 31;
  const int nt = (d >> 14) & 1, tr = (d >> 15) & 1, ab = (d >> 16) & 1;
  const int fin = tr ? C : R, fout = tr ? R : C;
  const float* W = mats[src];
  const int n = nt * 16 + (lane & 15);
  const int kb = (lane < 16) ? 0 : 8;
  for (int j = 0; j < 16; ++j) {
    const int k = kb + j + ((j >= 8) ? 8 : 0);
    float v = 0.f;
    if (k < fin && n < fout) v = tr ? W[n * C + k] : W[k * C + n];
    if (ab) v = fabsf(v);
    WF[tile * 512 + lane * 16 + j] = (_Float16)v;
  }
}

__global__ void pack_params(NetParams P, float* __restrict__ PAR) {
  const int i = threadIdx.x;
  if (i >= PAR_N) return;
  float v = 0.f;
  if (i < 32)                          { if (i < 30) v = P.z1_b[i]; }
  else if (i < 64)                     { const int k = i - 32;  if (k < 30) v = P.e_b0[k]; }
  else if (i < 96)                     { const int k = i - 64;  if (k < 30) v = P.e_b1[k]; }
  else if (i < 128)                    { const int k = i - 96;  if (k < 30) v = P.e_b2[k]; }
  else if (i == 128)                   { v = P.e_b3[0]; }
  else if (i >= P_A4V && i < P_A4V+32) { const int k = i - P_A4V; if (k < 30) v = fabsf(P.z4[k]); }
  else if (i >= P_Q4V && i < P_Q4V+32) { const int k = i - P_Q4V; if (k < 30) v = P.q4[k]; }
  else if (i >= P_E3V && i < P_E3V+32) { const int k = i - P_E3V; if (k < 30) v = P.e_w3[k]; }
  else if (i >= P_SZ3 && i < P_SZ3+4)  { v = P.sz3[i - P_SZ3]; }
  PAR[i] = v;
}

// ---------------- main kernel shared state ----------------
struct __align__(32) SH {
  _Float16 tS[512];      // network input tile (s / ev-x)
  _Float16 tP[512];      // peq input tile
  _Float16 tA[512];      // activation / u ping
  _Float16 tB[512];      // activation / u pong
  _Float16 mk[6][512];   // derivative-mask tiles
  float par[PAR_N];
  float fy[16], gq[16], tfy[16], tgq[16], hv[16], hq[16];
  float colA[16], colB[16];
  float gs[64], gpe[64], tgs[64], tgpe[64], hs[64], hpe[64];
  float rs[256];         // [m][i][j] Hessian d g_s_i / d s_j
  float ST[64], PE0[64], Q0[16], PEn[64], Qn[16], dlam[16];
  int msk[16];
};

__device__ __forceinline__ v8f vzero() {
  v8f z;
  #pragma unroll
  for (int i = 0; i < 8; ++i) z[i] = 0.f;
  return z;
}

// A-fragment (16x32 f16): lanes 0-15 -> M=lane, K in {0..7,16..23};
// lanes 16-31 -> M=lane-16, K in {8..15,24..31}.  (ISA 7.12.2)
__device__ __forceinline__ v16h load_a(const _Float16* tile, int lane) {
  const int m = lane & 15;
  const int kb = (lane < 16) ? 0 : 8;
  const v8h lo = *(const v8h*)(tile + m * 32 + kb);
  const v8h hi = *(const v8h*)(tile + m * 32 + 16 + kb);
  v16h r;
  #pragma unroll
  for (int i = 0; i < 8; ++i) { r[i] = lo[i]; r[i + 8] = hi[i]; }
  return r;
}

// Weight-fragment load, forced into the global address space (addrspace(1))
// so it lowers to global_load_b128 (LOADcnt only) instead of flat_load_b128
// (which would also tick DScnt and serialize against the LDS tile traffic).
typedef __attribute__((address_space(1))) const v16h gc_v16h;

__device__ __forceinline__ v16h ldb(const _Float16* __restrict__ WF, int tile, int lane) {
  const unsigned long long a =
      (unsigned long long)(const void*)(WF + (size_t)tile * 512 + (size_t)lane * 16);
  return *(gc_v16h*)a;
}

__device__ __forceinline__ v8f wmma(v16h a, v16h b, v8f c) {
  return __builtin_amdgcn_wmma_f32_16x16x32_f16(false, a, false, b, (short)0, c, false, false);
}

// C-frag (m = r + 8*(lane>=16), n = ntile*16 + (lane&15)) -> LDS tile,
// fused bias + activation, derivative mask stored alongside.
// act: 1 = elu, else leaky_relu(0.01)
__device__ __forceinline__ void store_act(_Float16* out, _Float16* dm, v8f c, int lane,
                                          int ntile, const float* bias, int act) {
  const int n = ntile * 16 + (lane & 15);
  const int mb = (lane < 16) ? 0 : 8;
  const float bv = bias ? bias[n] : 0.f;
  #pragma unroll
  for (int r = 0; r < 8; ++r) {
    const float z = c[r] + bv;
    float a, d;
    if (act == 1) { const float e = __expf(z); a = (z > 0.f) ? z : (e - 1.f); d = (z > 0.f) ? 1.f : e; }
    else          { a = (z > 0.f) ? z : 0.01f * z; d = (z > 0.f) ? 1.f : 0.01f; }
    out[(mb + r) * 32 + n] = (_Float16)a;
    dm[(mb + r) * 32 + n]  = (_Float16)d;
  }
}

// backprop: store (C-frag * stored mask) as next u-tile
__device__ __forceinline__ void store_mul(_Float16* out, const _Float16* dm, v8f c,
                                          int lane, int ntile) {
  const int n = ntile * 16 + (lane & 15);
  const int mb = (lane < 16) ? 0 : 8;
  #pragma unroll
  for (int r = 0; r < 8; ++r)
    out[(mb + r) * 32 + n] = (_Float16)(c[r] * (float)dm[(mb + r) * 32 + n]);
}

__device__ __forceinline__ void store_col0(float* out16, v8f c, int lane, float bias) {
  if ((lane & 15) == 0) {
    const int mb = (lane < 16) ? 0 : 8;
    #pragma unroll
    for (int r = 0; r < 8; ++r) out16[mb + r] = c[r] + bias;
  }
}

// u3 = head_vector (*) mask, broadcast over rows  (tile built by whole wave)
__device__ __forceinline__ void build_u(_Float16* t, const float* vec, const _Float16* mk,
                                        int lane, float sgn) {
  #pragma unroll
  for (int i = 0; i < 16; ++i) {
    const int e = lane * 16 + i;
    const int k = e & 31;
    t[e] = (_Float16)(sgn * vec[k] * (float)mk[e]);
  }
}

// ---------------- yield net: value + gradients (one eval) ----------------
__device__ __attribute__((noinline))
void fy_grad(const _Float16* __restrict__ WF, SH* sh, int lane,
             const float* S, const float* PE, const float* Q,
             int pd, float eps,
             float* o_fy, float* o_gs, float* o_gq, float* o_gpe) {
  // build input tiles: s (optionally perturbed along dim pd) and [pe,q]
  #pragma unroll
  for (int i = 0; i < 16; ++i) {
    const int e = lane * 16 + i;
    const int m = e >> 5, k = e & 31;
    const float sv = (k < 4) ? (S[m * 4 + k] + ((k == pd) ? eps : 0.f)) : 0.f;
    sh->tS[e] = (_Float16)sv;
    float pv = 0.f;
    if (k < 4) pv = PE[m * 4 + k]; else if (k == 4) pv = Q[m];
    sh->tP[e] = (_Float16)pv;
  }
  __syncthreads();
  const v16h aS = load_a(sh->tS, lane);
  const v8f zero = vzero();
  v8f c0, c1;
  // z1 = elu(s@W1 + b1)
  c0 = wmma(aS, ldb(WF, T_W1_0, lane), zero);
  c1 = wmma(aS, ldb(WF, T_W1_1, lane), zero);
  store_act(sh->tA, sh->mk[0], c0, lane, 0, sh->par + P_Z1B, 1);
  store_act(sh->tA, sh->mk[0], c1, lane, 1, sh->par + P_Z1B, 1);
  __syncthreads();
  { // z2 = elu(z1@|A2| + s@S1)
    const v16h a1 = load_a(sh->tA, lane);
    c0 = wmma(a1, ldb(WF, T_A2_0, lane), wmma(aS, ldb(WF, T_S1_0, lane), zero));
    c1 = wmma(a1, ldb(WF, T_A2_1, lane), wmma(aS, ldb(WF, T_S1_1, lane), zero));
    store_act(sh->tB, sh->mk[1], c0, lane, 0, nullptr, 1);
    store_act(sh->tB, sh->mk[1], c1, lane, 1, nullptr, 1);
  }
  __syncthreads();
  { // z3 = elu(z2@|A3| + s@S2)
    const v16h a2 = load_a(sh->tB, lane);
    c0 = wmma(a2, ldb(WF, T_A3_0, lane), wmma(aS, ldb(WF, T_S2_0, lane), zero));
    c1 = wmma(a2, ldb(WF, T_A3_1, lane), wmma(aS, ldb(WF, T_S2_1, lane), zero));
    store_act(sh->tA, sh->mk[2], c0, lane, 0, nullptr, 1);
    store_act(sh->tA, sh->mk[2], c1, lane, 1, nullptr, 1);
  }
  __syncthreads();
  { // z4 = z3@|A4| + s@S3  (scalar column)
    const v16h a3 = load_a(sh->tA, lane);
    const v8f cz = wmma(a3, ldb(WF, T_A4_0, lane), wmma(aS, ldb(WF, T_S3_0, lane), zero));
    store_col0(sh->colA, cz, lane, 0.f);
  }
  { // q branch: y1 = lrelu(peq@Q1)
    const v16h aP = load_a(sh->tP, lane);
    c0 = wmma(aP, ldb(WF, T_Q1_0, lane), zero);
    c1 = wmma(aP, ldb(WF, T_Q1_1, lane), zero);
    store_act(sh->tB, sh->mk[3], c0, lane, 0, nullptr, 2);
    store_act(sh->tB, sh->mk[3], c1, lane, 1, nullptr, 2);
  }
  __syncthreads();
  {
    const v16h y1 = load_a(sh->tB, lane);
    c0 = wmma(y1, ldb(WF, T_Q2_0, lane), zero);
    c1 = wmma(y1, ldb(WF, T_Q2_1, lane), zero);
    store_act(sh->tA, sh->mk[4], c0, lane, 0, nullptr, 2);
    store_act(sh->tA, sh->mk[4], c1, lane, 1, nullptr, 2);
  }
  __syncthreads();
  {
    const v16h y2 = load_a(sh->tA, lane);
    c0 = wmma(y2, ldb(WF, T_Q3_0, lane), zero);
    c1 = wmma(y2, ldb(WF, T_Q3_1, lane), zero);
    store_act(sh->tB, sh->mk[5], c0, lane, 0, nullptr, 2);
    store_act(sh->tB, sh->mk[5], c1, lane, 1, nullptr, 2);
  }
  __syncthreads();
  {
    const v16h y3 = load_a(sh->tB, lane);
    const v8f cq = wmma(y3, ldb(WF, T_Q4_0, lane), zero);
    store_col0(sh->colB, cq, lane, 0.f);
  }
  __syncthreads();
  if (lane < 16) o_fy[lane] = sh->colA[lane] - sh->colB[lane];
  // ---- backward, z branch: g_s ----
  build_u(sh->tA, sh->par + P_A4V, sh->mk[2], lane, 1.f);   // u3p
  __syncthreads();
  v8f gsa;
  { const int n = lane & 15;
    const float iv = (n < 4) ? sh->par[P_SZ3 + n] : 0.f;    // + S3 constant term
    #pragma unroll
    for (int r = 0; r < 8; ++r) gsa[r] = iv; }
  {
    const v16h u3 = load_a(sh->tA, lane);
    gsa = wmma(u3, ldb(WF, T_S2T_0, lane), gsa);
    c0 = wmma(u3, ldb(WF, T_A3T_0, lane), zero);
    c1 = wmma(u3, ldb(WF, T_A3T_1, lane), zero);
    store_mul(sh->tB, sh->mk[1], c0, lane, 0);
    store_mul(sh->tB, sh->mk[1], c1, lane, 1);
  }
  __syncthreads();
  {
    const v16h u2 = load_a(sh->tB, lane);
    gsa = wmma(u2, ldb(WF, T_S1T_0, lane), gsa);
    c0 = wmma(u2, ldb(WF, T_A2T_0, lane), zero);
    c1 = wmma(u2, ldb(WF, T_A2T_1, lane), zero);
    store_mul(sh->tA, sh->mk[0], c0, lane, 0);
    store_mul(sh->tA, sh->mk[0], c1, lane, 1);
  }
  __syncthreads();
  {
    const v16h u1 = load_a(sh->tA, lane);
    gsa = wmma(u1, ldb(WF, T_W1T_0, lane), gsa);
  }
  { const int n = lane & 15, mb = (lane < 16) ? 0 : 8;
    if (n < 4) {
      #pragma unroll
      for (int r = 0; r < 8; ++r) o_gs[(mb + r) * 4 + n] = gsa[r];
    } }
  // ---- backward, q branch: g_pe, g_q  (f = z4 - q4 -> negative head) ----
  build_u(sh->tB, sh->par + P_Q4V, sh->mk[5], lane, -1.f);
  __syncthreads();
  {
    const v16h v3 = load_a(sh->tB, lane);
    c0 = wmma(v3, ldb(WF, T_Q3T_0, lane), zero);
    c1 = wmma(v3, ldb(WF, T_Q3T_1, lane), zero);
    store_mul(sh->tA, sh->mk[4], c0, lane, 0);
    store_mul(sh->tA, sh->mk[4], c1, lane, 1);
  }
  __syncthreads();
  {
    const v16h v2 = load_a(sh->tA, lane);
    c0 = wmma(v2, ldb(WF, T_Q2T_0, lane), zero);
    c1 = wmma(v2, ldb(WF, T_Q2T_1, lane), zero);
    store_mul(sh->tB, sh->mk[3], c0, lane, 0);
    store_mul(sh->tB, sh->mk[3], c1, lane, 1);
  }
  __syncthreads();
  {
    const v16h v1 = load_a(sh->tB, lane);
    const v8f gpq = wmma(v1, ldb(WF, T_Q1T_0, lane), zero);
    const int n = lane & 15, mb = (lane < 16) ? 0 : 8;
    if (n < 4) {
      #pragma unroll
      for (int r = 0; r < 8; ++r) o_gpe[(mb + r) * 4 + n] = gpq[r];
    } else if (n == 4) {
      #pragma unroll
      for (int r = 0; r < 8; ++r) o_gq[mb + r] = gpq[r];
    }
  }
  __syncthreads();
}

// yield: value + grads + FD Hessian r_s[i][j] = d g_s_i / d s_j
__device__ void fy_full(const _Float16* __restrict__ WF, SH* sh, int lane,
                        const float* S, const float* PE, const float* Q) {
  fy_grad(WF, sh, lane, S, PE, Q, -1, 0.f, sh->fy, sh->gs, sh->gq, sh->gpe);
  __syncthreads();
  for (int d = 0; d < 4; ++d) {
    fy_grad(WF, sh, lane, S, PE, Q, d, FD_EPS, sh->tfy, sh->tgs, sh->tgq, sh->tgpe);
    __syncthreads();
    if (lane < 16) {
      const float inv = 1.0f / FD_EPS;
      for (int i = 0; i < 4; ++i)
        sh->rs[lane * 16 + i * 4 + d] = (sh->tgs[lane * 4 + i] - sh->gs[lane * 4 + i]) * inv;
    }
    __syncthreads();
  }
}

// ---------------- evolution net: value + gradients ----------------
__device__ __attribute__((noinline))
void ev_grad(const _Float16* __restrict__ WF, SH* sh, int lane,
             const float* S, const float* PE, const float* Q) {
  #pragma unroll
  for (int i = 0; i < 16; ++i) {
    const int e = lane * 16 + i;
    const int m = e >> 5, k = e & 31;
    float v = 0.f;
    if (k < 4) v = PE[m * 4 + k];
    else if (k == 4) v = Q[m];
    else if (k < 9) v = S[m * 4 + (k - 5)];
    sh->tS[e] = (_Float16)v;
  }
  __syncthreads();
  const v16h aX = load_a(sh->tS, lane);
  const v8f zero = vzero();
  v8f c0, c1;
  c0 = wmma(aX, ldb(WF, T_E0_0, lane), zero);
  c1 = wmma(aX, ldb(WF, T_E0_1, lane), zero);
  store_act(sh->tA, sh->mk[0], c0, lane, 0, sh->par + P_EB0, 2);
  store_act(sh->tA, sh->mk[0], c1, lane, 1, sh->par + P_EB0, 2);
  __syncthreads();
  {
    const v16h h0 = load_a(sh->tA, lane);
    c0 = wmma(h0, ldb(WF, T_E1_0, lane), zero);
    c1 = wmma(h0, ldb(WF, T_E1_1, lane), zero);
    store_act(sh->tB, sh->mk[1], c0, lane, 0, sh->par + P_EB1, 2);
    store_act(sh->tB, sh->mk[1], c1, lane, 1, sh->par + P_EB1, 2);
  }
  __syncthreads();
  {
    const v16h h1 = load_a(sh->tB, lane);
    c0 = wmma(h1, ldb(WF, T_E2_0, lane), zero);
    c1 = wmma(h1, ldb(WF, T_E2_1, lane), zero);
    store_act(sh->tA, sh->mk[2], c0, lane, 0, sh->par + P_EB2, 2);
    store_act(sh->tA, sh->mk[2], c1, lane, 1, sh->par + P_EB2, 2);
  }
  __syncthreads();
  {
    const v16h h2 = load_a(sh->tA, lane);
    const v8f ch = wmma(h2, ldb(WF, T_E3_0, lane), zero);
    store_col0(sh->colA, ch, lane, sh->par[P_EB3]);
  }
  __syncthreads();
  if (lane < 16) sh->hv[lane] = sh->colA[lane];
  // backward
  build_u(sh->tB, sh->par + P_E3V, sh->mk[2], lane, 1.f);
  __syncthreads();
  {
    const v16h t2 = load_a(sh->tB, lane);
    c0 = wmma(t2, ldb(WF, T_E2T_0, lane), zero);
    c1 = wmma(t2, ldb(WF, T_E2T_1, lane), zero);
    store_mul(sh->tA, sh->mk[1], c0, lane, 0);
    store_mul(sh->tA, sh->mk[1], c1, lane, 1);
  }
  __syncthreads();
  {
    const v16h t1 = load_a(sh->tA, lane);
    c0 = wmma(t1, ldb(WF, T_E1T_0, lane), zero);
    c1 = wmma(t1, ldb(WF, T_E1T_1, lane), zero);
    store_mul(sh->tB, sh->mk[0], c0, lane, 0);
    store_mul(sh->tB, sh->mk[0], c1, lane, 1);
  }
  __syncthreads();
  {
    const v16h t0 = load_a(sh->tB, lane);
    const v8f g = wmma(t0, ldb(WF, T_E0T_0, lane), zero);
    const int n = lane & 15, mb = (lane < 16) ? 0 : 8;
    if (n < 4) {
      #pragma unroll
      for (int r = 0; r < 8; ++r) sh->hpe[(mb + r) * 4 + n] = g[r];
    } else if (n == 4) {
      #pragma unroll
      for (int r = 0; r < 8; ++r) sh->hq[mb + r] = g[r];
    } else if (n < 9) {
      #pragma unroll
      for (int r = 0; r < 8; ++r) sh->hs[(mb + r) * 4 + (n - 5)] = g[r];
    }
  }
  __syncthreads();
}

// ---------------- per-element Newton update (lane m < 16) ----------------
__device__ void newton_update(SH* sh, int m) {
  // C_INV in closed form for E=80, nu=0.25
  const float CI[4][4] = {
    { 0.0125f,    -0.003125f, 0.f,      -0.003125f},
    {-0.003125f,  0.0125f,    0.f,      -0.003125f},
    { 0.f,        0.f,        0.03125f,  0.f      },
    {-0.003125f, -0.003125f,  0.f,       0.0125f  }};
  const float dl0 = sh->dlam[m];
  const bool mk = sh->msk[m] != 0;
  float M[5][10];
  for (int i = 0; i < 5; ++i)
    for (int j = 0; j < 10; ++j) M[i][j] = 0.f;
  for (int i = 0; i < 4; ++i)
    for (int j = 0; j < 4; ++j) M[i][j] = CI[i][j] + dl0 * sh->rs[m * 16 + i * 4 + j];
  for (int j = 0; j < 4; ++j) {
    float hp = 0.f;
    for (int k = 0; k < 4; ++k) hp += sh->hpe[m * 4 + k] * CI[k][j];
    M[4][j] = dl0 * (sh->hs[m * 4 + j] - hp);
  }
  M[4][4] = -1.f + dl0 * sh->hq[m];
  for (int i = 0; i < 5; ++i) M[i][5 + i] = 1.f;
  // Gauss-Jordan inverse
  for (int p = 0; p < 5; ++p) {
    float piv = M[p][p];
    piv = (fabsf(piv) < 1e-20f) ? ((piv < 0.f) ? -1e-20f : 1e-20f) : piv;
    const float ip = 1.f / piv;
    for (int j = 0; j < 10; ++j) M[p][j] *= ip;
    for (int i = 0; i < 5; ++i)
      if (i != p) {
        const float f = M[i][p];
        for (int j = 0; j < 10; ++j) M[i][j] -= f * M[p][j];
      }
  }
  float r_[5], a_[5], fd[5];
  for (int i = 0; i < 4; ++i) {
    r_[i] = sh->gs[m * 4 + i];
    a_[i] = -sh->PEn[m * 4 + i] + sh->PE0[m * 4 + i] + dl0 * r_[i];
  }
  r_[4] = sh->hv[m];
  a_[4] = -sh->Qn[m] + sh->Q0[m] + dl0 * sh->hv[m];
  for (int j = 0; j < 4; ++j) {
    float fp = 0.f;
    for (int k = 0; k < 4; ++k) fp += sh->gpe[m * 4 + k] * CI[k][j];
    fd[j] = sh->gs[m * 4 + j] - fp;
  }
  fd[4] = sh->gq[m];
  float fdA[5];
  for (int j = 0; j < 5; ++j) {
    float s = 0.f;
    for (int i = 0; i < 5; ++i) s += fd[i] * M[i][5 + j];
    fdA[j] = s;
  }
  float num = sh->fy[m], den = 1.f;
  for (int j = 0; j < 5; ++j) num -= fdA[j] * a_[j];
  if (mk) { den = 0.f; for (int j = 0; j < 5; ++j) den += fdA[j] * r_[j]; }
  const float dl = num / den;
  float dsq[5];
  for (int i = 0; i < 5; ++i) {
    float sa = 0.f, sr = 0.f;
    for (int j = 0; j < 5; ++j) { sa += M[i][5 + j] * a_[j]; sr += M[i][5 + j] * r_[j]; }
    dsq[i] = -sa - dl * sr;
  }
  if (mk) {
    for (int i = 0; i < 4; ++i) {
      float dpe = 0.f;
      for (int j = 0; j < 4; ++j) dpe -= CI[i][j] * dsq[j];
      sh->PEn[m * 4 + i] += dpe;
      sh->ST[m * 4 + i]  += dsq[i];
    }
    sh->Qn[m] += dsq[4];
    sh->dlam[m] = dl0 + dl;
  }
}

// ---------------- main scan kernel: one wave = 16 batch elements ----------
__global__ __launch_bounds__(32) void plastic_scan_kernel(
    const float* __restrict__ dS, const _Float16* __restrict__ WF,
    const float* __restrict__ PAR, float* __restrict__ out, int Bn, int T) {
  __shared__ SH sh;
  const int lane = threadIdx.x;
  const int e0 = blockIdx.x * 16;
  for (int i = lane; i < PAR_N; i += 32) sh.par[i] = PAR[i];
  if (lane < 16) {
    const int m = lane;
    const size_t b = (size_t)(e0 + m);
    for (int c = 0; c < 4; ++c) { sh.ST[m * 4 + c] = 0.f; sh.PE0[m * 4 + c] = 0.f; }
    sh.Q0[m] = 0.f;
    for (int c = 0; c < 4; ++c) out[(size_t)c * Bn * T + b * T] = 0.f;   // t = 0
  }
  __syncthreads();
  for (int t = 1; t < T; ++t) {
    if (lane < 16) {
      const int m = lane;
      const size_t b = (size_t)(e0 + m);
      for (int c = 0; c < 4; ++c) {
        sh.ST[m * 4 + c] += dS[b * T * 4 + (size_t)t * 4 + c];
        sh.PEn[m * 4 + c] = sh.PE0[m * 4 + c];
      }
      sh.Qn[m] = sh.Q0[m];
      sh.dlam[m] = 0.f;
    }
    __syncthreads();
    fy_full(WF, &sh, lane, sh.ST, sh.PE0, sh.Q0);
    __syncthreads();
    if (lane < 16) sh.msk[lane] = (sh.fy[lane] > 1e-6f) ? 1 : 0;
    __syncthreads();
    for (int it = 0; it < 2; ++it) {
      ev_grad(WF, &sh, lane, sh.ST, sh.PEn, sh.Qn);
      __syncthreads();
      if (lane < 16) newton_update(&sh, lane);
      __syncthreads();
      fy_full(WF, &sh, lane, sh.ST, sh.PEn, sh.Qn);
      __syncthreads();
      if (lane < 16) sh.msk[lane] = (sh.fy[lane] > 1e-6f) ? 1 : 0;
      __syncthreads();
    }
    if (lane < 16) {
      const int m = lane;
      const size_t b = (size_t)(e0 + m);
      for (int c = 0; c < 4; ++c) {
        out[(size_t)c * Bn * T + b * T + t] = sh.ST[m * 4 + c];
        sh.PE0[m * 4 + c] = sh.PEn[m * 4 + c];
      }
      sh.Q0[m] = sh.Qn[m];
    }
    __syncthreads();
  }
}

// ---------------- host launch ----------------
extern "C" void kernel_launch(void* const* d_in, const int* in_sizes, int n_in,
                              void* d_out, int out_size, void* d_ws, size_t ws_size,
                              hipStream_t stream) {
  if (n_in < 22) return;
  const float* dS = (const float*)d_in[1];   // d_stress (B,T,4)
  NetParams P;
  P.z1_w = (const float*)d_in[2];  P.z1_b = (const float*)d_in[3];
  P.z2   = (const float*)d_in[4];  P.z3   = (const float*)d_in[5];
  P.z4   = (const float*)d_in[6];  P.sz1  = (const float*)d_in[7];
  P.sz2  = (const float*)d_in[8];  P.sz3  = (const float*)d_in[9];
  P.q1   = (const float*)d_in[10]; P.q2   = (const float*)d_in[11];
  P.q3   = (const float*)d_in[12]; P.q4   = (const float*)d_in[13];
  P.e_w0 = (const float*)d_in[14]; P.e_b0 = (const float*)d_in[15];
  P.e_w1 = (const float*)d_in[16]; P.e_b1 = (const float*)d_in[17];
  P.e_w2 = (const float*)d_in[18]; P.e_b2 = (const float*)d_in[19];
  P.e_w3 = (const float*)d_in[20]; P.e_b3 = (const float*)d_in[21];

  const int T = 64;
  const int Bn = in_sizes[1] / (T * 4);
  const size_t frag_bytes = (size_t)NUM_TILES * 512 * sizeof(_Float16);
  if (ws_size < frag_bytes + PAR_N * sizeof(float)) return;

  _Float16* WF = (_Float16*)d_ws;
  float* PAR = (float*)((char*)d_ws + frag_bytes);

  pack_frags<<<(NUM_TILES * 32 + 255) / 256, 256, 0, stream>>>(P, WF);
  pack_params<<<1, 256, 0, stream>>>(P, PAR);
  plastic_scan_kernel<<<Bn / 16, 32, 0, stream>>>(dS, WF, PAR, (float*)d_out, Bn, T);
}
